// RSLOGIC2Model_16595753632538
// MI455X (gfx1250) — matrix operand order repaired and checked by reading, not compiled, not run on hardware
//
#include <hip/hip_runtime.h>

#define NUM_USERS 100000
#define NUM_ITEMS 50000
#define EMBED_K   64
#define N_INT     2000000
#define BATCH     4096

typedef __bf16 bf16;
typedef __attribute__((ext_vector_type(16))) bf16  v16bf;
typedef __attribute__((ext_vector_type(8)))  bf16  v8bf;
typedef __attribute__((ext_vector_type(8)))  float v8f;

__device__ __forceinline__ v8f wmma_bf16(v16bf a, v16bf b, v8f c) {
  // D = A(16x32 bf16) * B(32x16 bf16) + C(16x16 f32)
  return __builtin_amdgcn_wmma_f32_16x16x32_bf16(
      /*neg_a=*/false, a, /*neg_b=*/false, b,
      /*c_mod=*/(short)0, c, /*reuse_a=*/false, /*reuse_b=*/false);
}

// Build a 16x32 bf16 A fragment for this lane from two 8-float runs
// (contiguous K order, matched by B: lane = row M (mod 16), kh = lane>>4 selects K runs
//  [16*kh .. 16*kh+7] and [16*kh+8 .. 16*kh+15] within the 32-wide K tile).
__device__ __forceinline__ v16bf make_afrag(const float* __restrict__ run0,
                                            const float* __restrict__ run1) {
  const float4* p0 = (const float4*)run0;
  const float4* p1 = (const float4*)run1;
  float4 x0 = p0[0], x1 = p0[1];
  float4 y0 = p1[0], y1 = p1[1];
  v16bf a;
  a[0]  = (bf16)x0.x; a[1]  = (bf16)x0.y; a[2]  = (bf16)x0.z; a[3]  = (bf16)x0.w;
  a[4]  = (bf16)x1.x; a[5]  = (bf16)x1.y; a[6]  = (bf16)x1.z; a[7]  = (bf16)x1.w;
  a[8]  = (bf16)y0.x; a[9]  = (bf16)y0.y; a[10] = (bf16)y0.z; a[11] = (bf16)y0.w;
  a[12] = (bf16)y1.x; a[13] = (bf16)y1.y; a[14] = (bf16)y1.z; a[15] = (bf16)y1.w;
  return a;
}

// One wave processes one tile of 16 interactions, full N=64:
//   H   = leakyrelu(concat(Gu[u],Gi[i]) @ W1^T + b1)   (K=128, 16 WMMA)
//   OUT = H @ W2^T + b2                                (K=64,   8 WMMA)
// B1 (W1^T) fragments live in registers for the wave's lifetime (128 VGPRs,
// serves 2/3 of the FLOPs). B2 (W2^T) fragments are staged in LDS once per
// workgroup and re-read per tile (keeps the wave under 256 VGPRs -> no
// s_set_vgpr_msb churn, better occupancy for hiding L2 gather latency).
// ACCUM: atomically accumulate OUT into per-user sums + counts (segment mean).
// !ACCUM: write OUT rows to outbuf (batch gui pass).
template<bool ACCUM>
__global__ void __launch_bounds__(128)
repr_kernel(const float* __restrict__ Gu, const float* __restrict__ Gi,
            const bf16* __restrict__ W1bf, const bf16* __restrict__ W2bf,
            const float* __restrict__ b1,  const float* __restrict__ b2,
            const int* __restrict__ uptr,  const int* __restrict__ iptr,
            float* __restrict__ sums, float* __restrict__ counts,
            float* __restrict__ outbuf, int numTiles)
{
  __shared__ __align__(32) bf16 Hs[4][16 * EMBED_K];     // 2KB per wave
  __shared__ __align__(32) bf16 W2s[EMBED_K * EMBED_K];  // 8KB per workgroup

  const int lane = threadIdx.x & 31;
  const int wid  = threadIdx.x >> 5;
  const int ln   = lane & 15;            // row (A) or column (B/D) index
  const int kh   = lane >> 4;            // K-half selector
  const int kh8  = kh * 8;
  const int kh16 = kh * 16;

  // Stage W2^T into LDS (vectorized, once per workgroup).
  {
    const uint4* src = (const uint4*)W2bf;
    uint4*       dst = (uint4*)W2s;
    for (int j = threadIdx.x; j < (EMBED_K * EMBED_K) / 8; j += blockDim.x)
      dst[j] = src[j];
  }
  __syncthreads();

  // B1 = W1^T (128x64): B[k][n] = W1[n][k]; lane reads 16 contiguous k's.
  v16bf B1f[4][4];
#pragma unroll
  for (int kb = 0; kb < 4; ++kb)
#pragma unroll
    for (int nb = 0; nb < 4; ++nb)
      B1f[kb][nb] = *(const v16bf*)(W1bf + (nb * 16 + ln) * 128 + kb * 32 + kh16);

  float bias1[4], bias2[4];
#pragma unroll
  for (int nb = 0; nb < 4; ++nb) {
    bias1[nb] = b1[nb * 16 + ln];
    bias2[nb] = b2[nb * 16 + ln];
  }

  bf16* hs = &Hs[wid][0];

  const int  wavesPerBlock = blockDim.x >> 5;
  long       t       = (long)blockIdx.x * wavesPerBlock + wid;
  const long tstride = (long)gridDim.x * wavesPerBlock;

  for (; t < numTiles; t += tstride) {
    const int base = (int)t * 16;
    const int row  = base + ln;
    const int uid  = uptr[row];
    const int iid  = iptr[row];
    const float* urow = Gu + (long)uid * EMBED_K;
    const float* irow = Gi + (long)iid * EMBED_K;

    int uidr[8];
    if constexpr (ACCUM) {
#pragma unroll
      for (int r = 0; r < 8; ++r) uidr[r] = uptr[base + kh8 + r];
    }

    // ---- GEMM1: [16 x 128] @ [128 x 64], K-tiles 0,1 from Gu, 2,3 from Gi ----
    v8f C[4] = {v8f{}, v8f{}, v8f{}, v8f{}};
    {
      v16bf a = make_afrag(urow + kh16, urow + 8 + kh16);
#pragma unroll
      for (int nb = 0; nb < 4; ++nb) C[nb] = wmma_bf16(a, B1f[0][nb], C[nb]);
    }
    {
      v16bf a = make_afrag(urow + 32 + kh16, urow + 40 + kh16);
#pragma unroll
      for (int nb = 0; nb < 4; ++nb) C[nb] = wmma_bf16(a, B1f[1][nb], C[nb]);
    }
    {
      v16bf a = make_afrag(irow + kh16, irow + 8 + kh16);
#pragma unroll
      for (int nb = 0; nb < 4; ++nb) C[nb] = wmma_bf16(a, B1f[2][nb], C[nb]);
    }
    {
      v16bf a = make_afrag(irow + 32 + kh16, irow + 40 + kh16);
#pragma unroll
      for (int nb = 0; nb < 4; ++nb) C[nb] = wmma_bf16(a, B1f[3][nb], C[nb]);
    }

    // ---- epilogue 1: bias + leaky relu, re-layout D->A via LDS (bf16) ----
    // D layout: lane ln = column N, VGPR r = row M = r + 8*kh
#pragma unroll
    for (int nb = 0; nb < 4; ++nb) {
#pragma unroll
      for (int r = 0; r < 8; ++r) {
        float x = C[nb][r] + bias1[nb];
        x = x > 0.0f ? x : 0.01f * x;
        hs[(kh8 + r) * EMBED_K + nb * 16 + ln] = (bf16)x;
      }
    }
    // cross-lane exchange within the wave through LDS (DS ops are in-order per
    // wave; the wait + memory clobber pins compiler ordering)
    asm volatile("s_wait_dscnt 0" ::: "memory");

    // Opaque zero offset: makes W2 fragment addresses loop-variant so LICM
    // cannot hoist the 8 fragments back into 64 registers; the pointer still
    // derives from the __shared__ object, so these remain ds_load_b128.
    int w2off = 0;
    asm volatile("" : "+v"(w2off));
    const bf16* w2it = &W2s[w2off];

    // ---- GEMM2: [16 x 64] @ [64 x 64], B2 fragments streamed from LDS ----
    v8f D[4] = {v8f{}, v8f{}, v8f{}, v8f{}};
#pragma unroll
    for (int kb = 0; kb < 2; ++kb) {
      const bf16* hr = hs + ln * EMBED_K + kb * 32 + kh16;
      v8bf lo = *(const v8bf*)hr;          // 16B ds_load
      v8bf hi = *(const v8bf*)(hr + 8);    // 16B ds_load
      v16bf a2 = __builtin_shufflevector(lo, hi,
          0, 1, 2, 3, 4, 5, 6, 7, 8, 9, 10, 11, 12, 13, 14, 15);
#pragma unroll
      for (int nb = 0; nb < 4; ++nb) {
        v16bf bfrag = *(const v16bf*)(w2it + (nb * 16 + ln) * 64 + kb * 32 + kh16);
        D[nb] = wmma_bf16(a2, bfrag, D[nb]);
      }
    }

    // ---- epilogue 2: bias, then scatter-accumulate or store ----
#pragma unroll
    for (int nb = 0; nb < 4; ++nb) {
      const int n = nb * 16 + ln;
#pragma unroll
      for (int r = 0; r < 8; ++r) {
        float o = D[nb][r] + bias2[nb];
        if constexpr (ACCUM) {
          (void)__hip_atomic_fetch_add(sums + (long)uidr[r] * EMBED_K + n, o,
                                       __ATOMIC_RELAXED, __HIP_MEMORY_SCOPE_AGENT);
        } else {
          outbuf[(long)(base + kh8 + r) * EMBED_K + n] = o;
        }
      }
    }
    if constexpr (ACCUM) {
      if (lane < 16)  // rows 0..15 of the tile, once each
        (void)__hip_atomic_fetch_add(counts + uid, 1.0f,
                                     __ATOMIC_RELAXED, __HIP_MEMORY_SCOPE_AGENT);
    }
  }
}

__global__ void init_kernel(float* __restrict__ zbuf, long nzero,
                            const float* __restrict__ W1, const float* __restrict__ W2,
                            bf16* __restrict__ W1bf, bf16* __restrict__ W2bf)
{
  long i      = (long)blockIdx.x * blockDim.x + threadIdx.x;
  long stride = (long)gridDim.x * blockDim.x;
  for (long j = i; j < nzero; j += stride) zbuf[j] = 0.0f;
  for (long j = i; j < EMBED_K * 2 * EMBED_K; j += stride) W1bf[j] = (bf16)W1[j];
  for (long j = i; j < EMBED_K * EMBED_K; j += stride)     W2bf[j] = (bf16)W2[j];
}

// One wave per batch row: mean-divide, gamma_i gather, xui dot via shuffles.
__global__ void __launch_bounds__(128)
finalize_kernel(const float* __restrict__ sums, const float* __restrict__ counts,
                const float* __restrict__ gui,  const float* __restrict__ Gi,
                const int* __restrict__ users,  const int* __restrict__ items,
                float* __restrict__ out_xui, float* __restrict__ out_gustar,
                float* __restrict__ out_gammai)
{
  const int wid  = threadIdx.x >> 5;
  const int lane = threadIdx.x & 31;
  const int b    = blockIdx.x * 4 + wid;
  if (b >= BATCH) return;
  const int u  = users[b];
  const int it = items[b];
  const float inv = 1.0f / fmaxf(counts[u], 1.0f);
  float acc = 0.0f;
#pragma unroll
  for (int j = 0; j < 2; ++j) {
    const int n  = lane + j * 32;
    const float gs = sums[(long)u * EMBED_K + n] * inv;
    out_gustar[(long)b * EMBED_K + n] = gs;
    out_gammai[(long)b * EMBED_K + n] = Gi[(long)it * EMBED_K + n];
    acc += gs * gui[(long)b * EMBED_K + n];
  }
#pragma unroll
  for (int off = 16; off > 0; off >>= 1) acc += __shfl_down(acc, off, 32);
  if (lane == 0) out_xui[b] = acc;
}

extern "C" void kernel_launch(void* const* d_in, const int* in_sizes, int n_in,
                              void* d_out, int out_size, void* d_ws, size_t ws_size,
                              hipStream_t stream)
{
  const float* Gu    = (const float*)d_in[0];
  const float* Gi    = (const float*)d_in[1];
  const float* W1    = (const float*)d_in[2];
  const float* b1    = (const float*)d_in[3];
  const float* W2    = (const float*)d_in[4];
  const float* b2    = (const float*)d_in[5];
  const int*   users = (const int*)d_in[6];
  const int*   items = (const int*)d_in[7];
  const int*   ui    = (const int*)d_in[8];
  const int*   ui_u  = ui;
  const int*   ui_i  = ui + N_INT;

  // Workspace layout (~27 MB)
  float* sums   = (float*)d_ws;                         // [NUM_USERS * 64]
  float* counts = sums + (long)NUM_USERS * EMBED_K;     // [NUM_USERS]
  float* gui    = counts + NUM_USERS;                   // [BATCH * 64]
  bf16*  W1bf   = (bf16*)(gui + (long)BATCH * EMBED_K); // [64 * 128]
  bf16*  W2bf   = W1bf + EMBED_K * 2 * EMBED_K;         // [64 * 64]

  float* out_xui    = (float*)d_out;                    // [BATCH]
  float* out_gustar = out_xui + BATCH;                  // [BATCH * 64]
  float* out_gammai = out_gustar + (long)BATCH * EMBED_K; // [BATCH * 64]

  const long nzero = (long)NUM_USERS * EMBED_K + NUM_USERS;
  init_kernel<<<2048, 256, 0, stream>>>(sums, nzero, W1, W2, W1bf, W2bf);

  // 2M interactions -> 125000 tiles of 16
  repr_kernel<true><<<2048, 128, 0, stream>>>(Gu, Gi, W1bf, W2bf, b1, b2,
      ui_u, ui_i, sums, counts, nullptr, N_INT / 16);

  // batch pass -> 256 tiles of 16
  repr_kernel<false><<<64, 128, 0, stream>>>(Gu, Gi, W1bf, W2bf, b1, b2,
      users, items, nullptr, nullptr, gui, BATCH / 16);

  finalize_kernel<<<BATCH / 4, 128, 0, stream>>>(sums, counts, gui, Gi,
      users, items, out_xui, out_gustar, out_gammai);
}
